// GraphAwareAttention_19834158973472
// MI455X (gfx1250) — matrix operand order repaired and checked by reading, not compiled, not run on hardware
//
#include <hip/hip_runtime.h>
#include <hip/hip_bf16.h>

typedef __bf16 bf16_t;
typedef bf16_t v16bf __attribute__((ext_vector_type(16)));
typedef bf16_t v8bf  __attribute__((ext_vector_type(8)));   // 16 bytes -> ds b128
typedef float  v8f   __attribute__((ext_vector_type(8)));
typedef float  v4f   __attribute__((ext_vector_type(4)));
typedef int    i32x4 __attribute__((vector_size(16)));      // builtin's expected pointee
typedef __attribute__((address_space(1))) i32x4 gvec_t;     // global AS
typedef __attribute__((address_space(3))) i32x4 lvec_t;     // LDS AS

#define BATCH 2
#define SEQ   2048
#define HID   1024
#define NH    16
#define HD    64

#if defined(__has_builtin)
#if __has_builtin(__builtin_amdgcn_global_load_async_to_lds_b128) && \
    __has_builtin(__builtin_amdgcn_s_wait_asynccnt)
#define USE_ASYNC 1
#endif
#endif

// ---------------------------------------------------------------------------
// split-bf16 helpers: fp32 = hi(bf16) + lo(bf16 residual); product via 3 WMMAs
// ---------------------------------------------------------------------------
static __device__ __forceinline__ v8f wmma3(v16bf ah, v16bf al, v16bf bh, v16bf bl, v8f c) {
    c = __builtin_amdgcn_wmma_f32_16x16x32_bf16(false, ah, false, bh, (short)0, c, false, false);
    c = __builtin_amdgcn_wmma_f32_16x16x32_bf16(false, ah, false, bl, (short)0, c, false, false);
    c = __builtin_amdgcn_wmma_f32_16x16x32_bf16(false, al, false, bh, (short)0, c, false, false);
    return c;
}

static __device__ __forceinline__ void split16(const float* s, v16bf& hi, v16bf& lo) {
#pragma unroll
    for (int i = 0; i < 16; ++i) {
        float x = s[i];
        bf16_t hb = (bf16_t)x;
        hi[i] = hb;
        lo[i] = (bf16_t)(x - (float)hb);
    }
}

// load 8 fp32 from global, split, store hi/lo bf16 runs to LDS (16B each)
static __device__ __forceinline__ void stage_split8(const float* g, bf16_t* hi, bf16_t* lo) {
    v4f t0 = *(const v4f*)g, t1 = *(const v4f*)(g + 4);
    float r[8];
#pragma unroll
    for (int i = 0; i < 4; ++i) { r[i] = t0[i]; r[4 + i] = t1[i]; }
    v8bf h, l;
#pragma unroll
    for (int i = 0; i < 8; ++i) {
        bf16_t hb = (bf16_t)r[i];
        h[i] = hb;
        l[i] = (bf16_t)(r[i] - (float)hb);
    }
    *(v8bf*)hi = h;
    *(v8bf*)lo = l;
}

// A-operand fragment (16x32 bf16): lanes0-15 K={kb..kb+7, kb+16..kb+23}, kb=(lane>>4)*8 (+slab)
static __device__ __forceinline__ v16bf ldsA_frag(const bf16_t* rowbase, int kb) {
    union { v16bf v; v8bf h[2]; } u;
    u.h[0] = *(const v8bf*)(rowbase + kb);
    u.h[1] = *(const v8bf*)(rowbase + kb + 16);
    return u.v;
}
// B-operand fragment (32x16 bf16): K={kb..kb+7, kb+16..kb+23}, kb=slab*32+(lane>>4)*8
static __device__ __forceinline__ v16bf ldsB_frag(const bf16_t* rowbase, int kb) {
    union { v16bf v; v8bf h[2]; } u;
    u.h[0] = *(const v8bf*)(rowbase + kb);
    u.h[1] = *(const v8bf*)(rowbase + kb + 16);
    return u.v;
}

// ---------------------------------------------------------------------------
// GEMM: out[M,1024] = X[M,1024] @ W[1024,1024]^T + bias
// 128x128 block tile, 8 waves (4M x 2N), wave = 32x64, K slab = 32,
// operands split to hi/lo bf16 in LDS, 3x v_wmma_f32_16x16x32_bf16 per tile.
// ---------------------------------------------------------------------------
#define ASTR 40   // bf16 row stride: 80B rows, 16B aligned, conflict-free frag reads

__global__ __launch_bounds__(256) void gemm_bias_kernel(const float* __restrict__ X,
                                                        const float* __restrict__ W,
                                                        const float* __restrict__ bias,
                                                        float* __restrict__ out)
{
    __shared__ bf16_t Ahi[128 * ASTR], Alo[128 * ASTR];
    __shared__ bf16_t Bhi[128 * ASTR], Blo[128 * ASTR];

    const int tid  = threadIdx.x;
    const int lane = tid & 31;
    const int wave = tid >> 5;
    const int wm   = wave & 3;
    const int wn   = wave >> 2;
    const int m0   = blockIdx.y * 128;
    const int n0   = blockIdx.x * 128;
    const int kn   = lane & 15;
    const int hi8  = (lane >> 4) * 8;
    const int kbA  = (lane >> 4) * 8;
    const int kbB  = (lane >> 4) * 8;

    v8f acc[2][4];
#pragma unroll
    for (int i = 0; i < 2; ++i)
#pragma unroll
        for (int j = 0; j < 4; ++j)
#pragma unroll
            for (int r = 0; r < 8; ++r) acc[i][j][r] = 0.0f;

    const int lrow = tid >> 1;          // 0..127
    const int lcol = (tid & 1) * 16;    // 0 or 16
    const float* xp0 = X + (size_t)(m0 + lrow) * HID + lcol;
    const float* wp0 = W + (size_t)(n0 + lrow) * HID + lcol;

    for (int k0 = 0; k0 < HID; k0 += 32) {
        stage_split8(xp0 + k0,     Ahi + lrow * ASTR + lcol,     Alo + lrow * ASTR + lcol);
        stage_split8(xp0 + k0 + 8, Ahi + lrow * ASTR + lcol + 8, Alo + lrow * ASTR + lcol + 8);
        stage_split8(wp0 + k0,     Bhi + lrow * ASTR + lcol,     Blo + lrow * ASTR + lcol);
        stage_split8(wp0 + k0 + 8, Bhi + lrow * ASTR + lcol + 8, Blo + lrow * ASTR + lcol + 8);
        __syncthreads();

        v16bf ah[2], al[2];
#pragma unroll
        for (int mi = 0; mi < 2; ++mi) {
            const int rowA = (wm * 32 + mi * 16 + kn) * ASTR;
            ah[mi] = ldsA_frag(Ahi + rowA, kbA);
            al[mi] = ldsA_frag(Alo + rowA, kbA);
        }
#pragma unroll
        for (int ni = 0; ni < 4; ++ni) {
            const int rowB = (wn * 64 + ni * 16 + kn) * ASTR;
            v16bf bh = ldsB_frag(Bhi + rowB, kbB);
            v16bf bl = ldsB_frag(Blo + rowB, kbB);
#pragma unroll
            for (int mi = 0; mi < 2; ++mi)
                acc[mi][ni] = wmma3(ah[mi], al[mi], bh, bl, acc[mi][ni]);
        }
        __syncthreads();
    }

#pragma unroll
    for (int mi = 0; mi < 2; ++mi)
#pragma unroll
        for (int ni = 0; ni < 4; ++ni) {
            const int col = n0 + wn * 64 + ni * 16 + kn;
            const float bv = bias[col];
#pragma unroll
            for (int r = 0; r < 8; ++r) {
                const int row = m0 + wm * 32 + mi * 16 + r + hi8;
                out[(size_t)row * HID + col] = acc[mi][ni][r] + bv;
            }
        }
}

// ---------------------------------------------------------------------------
// Attention: block = (b,h, 128-q-tile), 8 waves x 16 q-rows, k-tiles of 32.
// Pass 1: online softmax stats. Pass 2: write normalized P (NT), O += P V.
// All matmuls via split-bf16 3x WMMA. sim tiles staged with async global->LDS.
// ---------------------------------------------------------------------------
#define KSTRD 72   // K tile [kcol 32][d 64] bf16
#define VSTRD 40   // V tile transposed [d 64][kcol 32] bf16
#define SSTRD 36   // sim tile [128][32] fp32
#define PSTRD 36   // per-wave P staging [16][32] fp32

__global__ __launch_bounds__(256) void attention_kernel(const float* __restrict__ Qb,
                                                        const float* __restrict__ Kb,
                                                        const float* __restrict__ Vb,
                                                        const int*   __restrict__ amask,
                                                        const float* __restrict__ sim,
                                                        const float* __restrict__ betaP,
                                                        float* __restrict__ attnW,
                                                        float* __restrict__ Ob)
{
    __shared__ bf16_t Khi[32 * KSTRD], Klo[32 * KSTRD];
    __shared__ bf16_t Vhi[64 * VSTRD], Vlo[64 * VSTRD];
    __shared__ float  Ss[128 * SSTRD];
    __shared__ float  Ps[8 * 16 * PSTRD];

    const int tid   = threadIdx.x;
    const int lane  = tid & 31;
    const int w     = tid >> 5;
    const int bh    = blockIdx.y;
    const int b     = bh >> 4;
    const int h     = bh & 15;
    const int qbase = blockIdx.x * 128;
    const int kn    = lane & 15;
    const int hi8   = (lane >> 4) * 8;
    const int kbA   = (lane >> 4) * 8;
    const int kbB   = (lane >> 4) * 8;
    const float beta  = betaP[0];
    const float scale = 0.125f;   // 1/sqrt(64)

    // Q A-fragments (hi/lo), 2 slabs of K=32 over HD=64, resident all kernel
    v16bf qhi[2], qlo[2];
    {
        const int qrow = qbase + w * 16 + kn;
        const float* qp = Qb + ((size_t)b * SEQ + qrow) * HID + h * HD;
#pragma unroll
        for (int s = 0; s < 2; ++s) {
            float buf[16];
            const int base = s * 32 + kbA;
#pragma unroll
            for (int e = 0; e < 8; ++e) { buf[e] = qp[base + e]; buf[8 + e] = qp[base + 16 + e]; }
            split16(buf, qhi[s], qlo[s]);
        }
    }

    float mrow[8], lrowv[8];
#pragma unroll
    for (int r = 0; r < 8; ++r) { mrow[r] = -3.0e38f; lrowv[r] = 0.0f; }

    const int krw  = tid >> 3;         // 0..31  K-tile row (kcol)
    const int kdc  = (tid & 7) * 8;    // 0..56  K-tile d group
    const int vcol = tid & 31;         // 0..31  V row (kcol)
    const int vdb  = (tid >> 5) * 8;   // 0..56  V d group
    const int srow = tid >> 1;         // 0..127 sim row
    const int scol = (tid & 1) * 16;   // sim col group

    // ---------------- pass 1: softmax statistics ----------------
    for (int kt = 0; kt < SEQ / 32; ++kt) {
        const int kcol0 = kt * 32;
        stage_split8(Kb + ((size_t)b * SEQ + kcol0 + krw) * HID + h * HD + kdc,
                     Khi + krw * KSTRD + kdc, Klo + krw * KSTRD + kdc);
        {
            const float* ssrc = sim + ((size_t)b * SEQ + qbase + srow) * SEQ + kcol0 + scol;
            float* sdst = Ss + srow * SSTRD + scol;
#ifdef USE_ASYNC
#pragma unroll
            for (int j = 0; j < 4; ++j)
                __builtin_amdgcn_global_load_async_to_lds_b128(
                    (gvec_t*)(ssrc + 4 * j), (lvec_t*)(sdst + 4 * j), 0, 0);
            __builtin_amdgcn_s_wait_asynccnt(0);
#else
#pragma unroll
            for (int j = 0; j < 4; ++j)
                *(v4f*)(sdst + 4 * j) = *(const v4f*)(ssrc + 4 * j);
#endif
        }
        __syncthreads();

        v8f sacc[2];
#pragma unroll
        for (int n2 = 0; n2 < 2; ++n2)
#pragma unroll
            for (int r = 0; r < 8; ++r) sacc[n2][r] = 0.0f;
#pragma unroll
        for (int s = 0; s < 2; ++s)
#pragma unroll
            for (int n2 = 0; n2 < 2; ++n2) {
                const int rowK = (n2 * 16 + kn) * KSTRD;
                v16bf bhf = ldsB_frag(Khi + rowK, s * 32 + kbB);
                v16bf blf = ldsB_frag(Klo + rowK, s * 32 + kbB);
                sacc[n2] = wmma3(qhi[s], qlo[s], bhf, blf, sacc[n2]);
            }

#pragma unroll
        for (int n2 = 0; n2 < 2; ++n2) {
            const int mv = amask[b * SEQ + kcol0 + n2 * 16 + kn];
#pragma unroll
            for (int r = 0; r < 8; ++r) {
                float s = sacc[n2][r] * scale + beta * Ss[(w * 16 + r + hi8) * SSTRD + n2 * 16 + kn];
                if (mv == 0) s = -1.0e30f;
                float vmax = s;
                vmax = fmaxf(vmax, __shfl_xor(vmax, 1, 32));
                vmax = fmaxf(vmax, __shfl_xor(vmax, 2, 32));
                vmax = fmaxf(vmax, __shfl_xor(vmax, 4, 32));
                vmax = fmaxf(vmax, __shfl_xor(vmax, 8, 32));
                const float m_new = fmaxf(mrow[r], vmax);
                float e = __expf(s - m_new);
                e += __shfl_xor(e, 1, 32);
                e += __shfl_xor(e, 2, 32);
                e += __shfl_xor(e, 4, 32);
                e += __shfl_xor(e, 8, 32);
                lrowv[r] = lrowv[r] * __expf(mrow[r] - m_new) + e;
                mrow[r] = m_new;
            }
        }
        __syncthreads();
    }

    float inv_l[8];
#pragma unroll
    for (int r = 0; r < 8; ++r) inv_l[r] = 1.0f / fmaxf(lrowv[r], 1e-30f);

    v8f oacc[4];
#pragma unroll
    for (int ni = 0; ni < 4; ++ni)
#pragma unroll
        for (int r = 0; r < 8; ++r) oacc[ni][r] = 0.0f;

    // ---------------- pass 2: write P, accumulate O = P V ----------------
    for (int kt = 0; kt < SEQ / 32; ++kt) {
        const int kcol0 = kt * 32;
        stage_split8(Kb + ((size_t)b * SEQ + kcol0 + krw) * HID + h * HD + kdc,
                     Khi + krw * KSTRD + kdc, Klo + krw * KSTRD + kdc);
        {   // V tile, transposed into [d][kcol]
            const float* vp = Vb + ((size_t)b * SEQ + kcol0 + vcol) * HID + h * HD + vdb;
            v4f t0 = *(const v4f*)vp, t1 = *(const v4f*)(vp + 4);
#pragma unroll
            for (int j = 0; j < 8; ++j) {
                float x = (j < 4) ? t0[j] : t1[j - 4];
                bf16_t hb = (bf16_t)x;
                Vhi[(vdb + j) * VSTRD + vcol] = hb;
                Vlo[(vdb + j) * VSTRD + vcol] = (bf16_t)(x - (float)hb);
            }
        }
        {
            const float* ssrc = sim + ((size_t)b * SEQ + qbase + srow) * SEQ + kcol0 + scol;
            float* sdst = Ss + srow * SSTRD + scol;
#ifdef USE_ASYNC
#pragma unroll
            for (int j = 0; j < 4; ++j)
                __builtin_amdgcn_global_load_async_to_lds_b128(
                    (gvec_t*)(ssrc + 4 * j), (lvec_t*)(sdst + 4 * j), 0, 0);
            __builtin_amdgcn_s_wait_asynccnt(0);
#else
#pragma unroll
            for (int j = 0; j < 4; ++j)
                *(v4f*)(sdst + 4 * j) = *(const v4f*)(ssrc + 4 * j);
#endif
        }
        __syncthreads();

        v8f sacc[2];
#pragma unroll
        for (int n2 = 0; n2 < 2; ++n2)
#pragma unroll
            for (int r = 0; r < 8; ++r) sacc[n2][r] = 0.0f;
#pragma unroll
        for (int s = 0; s < 2; ++s)
#pragma unroll
            for (int n2 = 0; n2 < 2; ++n2) {
                const int rowK = (n2 * 16 + kn) * KSTRD;
                v16bf bhf = ldsB_frag(Khi + rowK, s * 32 + kbB);
                v16bf blf = ldsB_frag(Klo + rowK, s * 32 + kbB);
                sacc[n2] = wmma3(qhi[s], qlo[s], bhf, blf, sacc[n2]);
            }

#pragma unroll
        for (int n2 = 0; n2 < 2; ++n2) {
            const int mv = amask[b * SEQ + kcol0 + n2 * 16 + kn];
#pragma unroll
            for (int r = 0; r < 8; ++r) {
                float s = sacc[n2][r] * scale + beta * Ss[(w * 16 + r + hi8) * SSTRD + n2 * 16 + kn];
                if (mv == 0) s = -1.0e30f;
                const float p = __expf(s - mrow[r]) * inv_l[r];
                const int qglob = qbase + w * 16 + r + hi8;
                __builtin_nontemporal_store(p, attnW + ((size_t)bh * SEQ + qglob) * SEQ + kcol0 + n2 * 16 + kn);
                Ps[(w * 16 + r + hi8) * PSTRD + n2 * 16 + kn] = p;
            }
        }
        __syncthreads();

        // P tile as A operand (split hi/lo), V as B operand
        float pbuf[16];
        const float* pb = Ps + (w * 16 + kn) * PSTRD;
#pragma unroll
        for (int e = 0; e < 8; ++e) { pbuf[e] = pb[kbA + e]; pbuf[8 + e] = pb[kbA + 16 + e]; }
        v16bf phi, plo;
        split16(pbuf, phi, plo);
#pragma unroll
        for (int ni = 0; ni < 4; ++ni) {
            const int rowV = (ni * 16 + kn) * VSTRD;
            v16bf bhf = ldsB_frag(Vhi + rowV, kbB);
            v16bf blf = ldsB_frag(Vlo + rowV, kbB);
            oacc[ni] = wmma3(phi, plo, bhf, blf, oacc[ni]);
        }
        __syncthreads();
    }

    // O in (b, s, h*HD+d) layout for the final projection
#pragma unroll
    for (int ni = 0; ni < 4; ++ni) {
        const int col = h * HD + ni * 16 + kn;
#pragma unroll
        for (int r = 0; r < 8; ++r) {
            const int qglob = qbase + w * 16 + r + hi8;
            Ob[((size_t)b * SEQ + qglob) * HID + col] = oacc[ni][r];
        }
    }
}

// ---------------------------------------------------------------------------
extern "C" void kernel_launch(void* const* d_in, const int* in_sizes, int n_in,
                              void* d_out, int out_size, void* d_ws, size_t ws_size,
                              hipStream_t stream)
{
    const float* query = (const float*)d_in[0];
    const float* key   = (const float*)d_in[1];
    const float* value = (const float*)d_in[2];
    const int*   amask = (const int*)  d_in[3];
    const float* sim   = (const float*)d_in[4];
    const float* Wq    = (const float*)d_in[5];
    const float* bq    = (const float*)d_in[6];
    const float* Wk    = (const float*)d_in[7];
    const float* bk    = (const float*)d_in[8];
    const float* Wv    = (const float*)d_in[9];
    const float* bv    = (const float*)d_in[10];
    const float* Wo    = (const float*)d_in[11];
    const float* bo    = (const float*)d_in[12];
    const float* beta  = (const float*)d_in[13];

    const size_t bufElems = (size_t)BATCH * SEQ * HID;   // 16 MB each
    float* Qb = (float*)d_ws;
    float* Kb = Qb + bufElems;
    float* Vb = Kb + bufElems;
    float* Ob = Vb + bufElems;

    float* out   = (float*)d_out;
    float* attnW = (float*)d_out + bufElems;

    const dim3 ggrid(HID / 128, (BATCH * SEQ) / 128);   // (8, 32)
    gemm_bias_kernel<<<ggrid, 256, 0, stream>>>(query, Wq, bq, Qb);
    gemm_bias_kernel<<<ggrid, 256, 0, stream>>>(key,   Wk, bk, Kb);
    gemm_bias_kernel<<<ggrid, 256, 0, stream>>>(value, Wv, bv, Vb);

    attention_kernel<<<dim3(SEQ / 128, BATCH * NH), 256, 0, stream>>>(
        Qb, Kb, Vb, amask, sim, beta, attnW, Ob);

    gemm_bias_kernel<<<ggrid, 256, 0, stream>>>(Ob, Wo, bo, out);
}
